// GNNDenseLayer_41266045779977
// MI455X (gfx1250) — matrix operand order repaired and checked by reading, not compiled, not run on hardware
//
#include <hip/hip_runtime.h>
#include <hip/hip_bf16.h>
#include <math.h>

// ---------- types ----------
typedef __attribute__((ext_vector_type(16))) _Float16 v16h;
typedef __attribute__((ext_vector_type(8)))  _Float16 v8h;
typedef __attribute__((ext_vector_type(8)))  float    v8f;

#define HDIM 128
#define VDIM 400
#define BDIM 2
#define BV   (BDIM * VDIM)            // 800
#define BVV  (BDIM * VDIM * VDIM)     // 320000
#define EPSV 1e-5f

// ---------- WMMA helpers (CDNA5 v_wmma_f32_16x16x32_f16) ----------

// A fragment (16x32 f16, M x K). ISA layout: lanes 0-15 hold M=0..15 with
// K 0-7 (elems 0..7) and K 16-23 (elems 8..15); lanes 16-31 hold same M with
// K 8-15 and K 24-31.  sm: row-major 16x128 halves.
__device__ __forceinline__ v16h load_a_frag(const _Float16* sm, int lane, int kbase) {
    const int m  = lane & 15;
    const int kh = (lane & 16) ? 8 : 0;
    const v8h lo = *(const v8h*)(sm + m * HDIM + kbase + kh);
    const v8h hi = *(const v8h*)(sm + m * HDIM + kbase + 16 + kh);
    v16h a;
#pragma unroll
    for (int i = 0; i < 8; ++i) { a[i] = lo[i]; a[8 + i] = hi[i]; }
    return a;
}

// B fragment (32x16 f16, K x N).  B(k,n) = W[o=n][k] for y = x @ W^T.
// ISA layout (matches A): lanes 0-15 hold N=lane with K 0-7 (elems 0..7) and
// K 16-23 (elems 8..15); lanes 16-31 hold N=lane-16 with K 8-15 and K 24-31.
__device__ __forceinline__ v16h load_b_frag(const _Float16* Wh, int o, int kbase, int lane) {
    const int kh = (lane & 16) ? 8 : 0;
    const v8h lo = *(const v8h*)(Wh + o * HDIM + kbase + kh);
    const v8h hi = *(const v8h*)(Wh + o * HDIM + kbase + 16 + kh);
    v16h b;
#pragma unroll
    for (int i = 0; i < 8; ++i) { b[i] = lo[i]; b[8 + i] = hi[i]; }
    return b;
}

__device__ __forceinline__ v8f wmma_f16(v16h a, v16h b, v8f c) {
    return __builtin_amdgcn_wmma_f32_16x16x32_f16(
        /*neg_a=*/false, a, /*neg_b=*/false, b,
        /*c_mod=*/(short)0, c, /*reuse_a=*/false, /*reuse_b=*/false);
}

// ---------- K0: convert 5 weight mats to f16, zero BN stat accumulators ----
__global__ void gnn_k_init(const float* __restrict__ Uw, const float* __restrict__ Vw,
                           const float* __restrict__ Aw, const float* __restrict__ Bw,
                           const float* __restrict__ Cw,
                           _Float16* __restrict__ Wh, float* __restrict__ stats) {
    const int idx = blockIdx.x * 256 + threadIdx.x;        // 0 .. 81919
    const float* src[5] = {Uw, Vw, Aw, Bw, Cw};
    const int g = idx >> 14;                               // /16384 (uniform per block)
    const int r = idx & 16383;
    Wh[idx] = (_Float16)src[g][r];
    if (idx < 512) stats[idx] = 0.0f;                      // xs, xq, es, eq (4x128)
}

// ---------- K1: Ux/Vx/Ax/Bx = x @ W^T + b via WMMA ----------
// grid = (50 m-tiles, 4 weights), block = 256 (8 waves, wave owns 16 out cols)
__global__ void gnn_k_lin4(const float* __restrict__ x, const _Float16* __restrict__ Wh,
                           const float* __restrict__ Ub, const float* __restrict__ Vb,
                           const float* __restrict__ Ab, const float* __restrict__ Bb,
                           float* __restrict__ xw4) {
    __shared__ alignas(16) _Float16 smX[16 * HDIM];
    const int mt = blockIdx.x, g = blockIdx.y;
    const int t = threadIdx.x, lane = t & 31, wave = t >> 5;

    { // stage 16x128 f32 -> f16 LDS (coalesced float4 loads)
        const int row = t >> 4, col = (t & 15) * 8;
        const float4* src = (const float4*)(x + (mt * 16 + row) * HDIM + col);
        const float4 a0 = src[0], a1 = src[1];
        _Float16* d = smX + row * HDIM + col;
        d[0] = (_Float16)a0.x; d[1] = (_Float16)a0.y; d[2] = (_Float16)a0.z; d[3] = (_Float16)a0.w;
        d[4] = (_Float16)a1.x; d[5] = (_Float16)a1.y; d[6] = (_Float16)a1.z; d[7] = (_Float16)a1.w;
    }
    __syncthreads();

    const int o = wave * 16 + (lane & 15);
    const _Float16* W = Wh + g * (HDIM * HDIM);
    v8f c = {};
#pragma unroll
    for (int kc = 0; kc < 4; ++kc)
        c = wmma_f16(load_a_frag(smX, lane, kc * 32), load_b_frag(W, o, kc * 32, lane), c);

    const float* biases[4] = {Ub, Vb, Ab, Bb};
    const float bias = biases[g][o];
    float* dst = xw4 + (size_t)g * BV * HDIM;
    const int m0 = mt * 16 + ((lane & 16) ? 8 : 0);
#pragma unroll
    for (int r = 0; r < 8; ++r) dst[(m0 + r) * HDIM + o] = c[r] + bias;
}

// ---------- K2: fused Ce GEMM + e_new + sigmoid-agg + BN partial stats ----
// grid = 800 blocks, one per (b,i); block = 256 threads = 8 waves.
__global__ void gnn_k_edge(const float* __restrict__ e, const _Float16* __restrict__ Cwh,
                           const float* __restrict__ Cb, const float* __restrict__ xw4,
                           float* __restrict__ e_new_out, float* __restrict__ x_new,
                           float* __restrict__ stats) {
    __shared__ alignas(16) _Float16 smE[16 * HDIM];
    const int bi = blockIdx.x;                    // b*400 + i
    const int b  = bi / VDIM;
    const int t = threadIdx.x, lane = t & 31, wave = t >> 5;
    const int h = wave * 16 + (lane & 15);        // this lane's output channel

    v16h bf[4];                                   // Cw B-fragments, live whole block
#pragma unroll
    for (int kc = 0; kc < 4; ++kc) bf[kc] = load_b_frag(Cwh, h, kc * 32, lane);

    const float* Ux = xw4;
    const float* Vx = xw4 + (size_t)1 * BV * HDIM;
    const float* Ax = xw4 + (size_t)2 * BV * HDIM;
    const float* Bx = xw4 + (size_t)3 * BV * HDIM;
    const float addc = Ax[bi * HDIM + h] + Cb[h];

    float agg = 0.f, esum = 0.f, esq = 0.f;
    const int jr0 = (lane & 16) ? 8 : 0;          // D-layout row base for this half-wave
    const size_t ebase = (size_t)bi * VDIM * HDIM;
    const int srow = t >> 4, scol = (t & 15) * 8; // staging coords

    for (int jt = 0; jt < 25; ++jt) {
        { // stage e tile (16 j-rows x 128) to LDS as f16
            const float4* src = (const float4*)(e + ebase + (size_t)(jt * 16 + srow) * HDIM + scol);
            const float4 a0 = src[0], a1 = src[1];
            if (jt < 24) __builtin_prefetch(src + (16 * HDIM) / 4, 0, 1);  // global_prefetch_b8
            _Float16* d = smE + srow * HDIM + scol;
            d[0] = (_Float16)a0.x; d[1] = (_Float16)a0.y; d[2] = (_Float16)a0.z; d[3] = (_Float16)a0.w;
            d[4] = (_Float16)a1.x; d[5] = (_Float16)a1.y; d[6] = (_Float16)a1.z; d[7] = (_Float16)a1.w;
        }
        __syncthreads();
        v16h af[4];
#pragma unroll
        for (int kc = 0; kc < 4; ++kc) af[kc] = load_a_frag(smE, lane, kc * 32);
        __syncthreads();                          // LDS free for next tile

        v8f c = {};
#pragma unroll
        for (int kc = 0; kc < 4; ++kc) c = wmma_f16(af[kc], bf[kc], c);

#pragma unroll
        for (int r = 0; r < 8; ++r) {
            const int j = jt * 16 + jr0 + r;
            const float bxv = Bx[(b * VDIM + j) * HDIM + h];
            const float val = c[r] + addc + bxv;                     // e_new
            e_new_out[ebase + (size_t)j * HDIM + h] = val;
            const float s = 1.0f / (1.0f + __expf(-val));            // sigmoid gate
            agg  += Vx[(b * VDIM + j) * HDIM + h] * s;
            esum += val;
            esq  += val * val;
        }
    }

    // lanes l and l^16 cover the same channel h with disjoint j's -> pair-reduce
    agg  += __shfl_xor(agg,  16, 32);
    esum += __shfl_xor(esum, 16, 32);
    esq  += __shfl_xor(esq,  16, 32);
    if ((lane & 16) == 0) {
        const float xn = Ux[bi * HDIM + h] + agg;
        x_new[bi * HDIM + h] = xn;
        atomicAdd(&stats[h],           xn);
        atomicAdd(&stats[HDIM + h],    xn * xn);
        atomicAdd(&stats[2 * HDIM + h], esum);
        atomicAdd(&stats[3 * HDIM + h], esq);
    }
}

// ---------- K3: batchnorm+relu for x ----------
__global__ void gnn_k_bn_x(const float* __restrict__ x_new, const float* __restrict__ stats,
                           const float* __restrict__ gx, const float* __restrict__ bx,
                           float* __restrict__ x_out) {
    const int idx = blockIdx.x * 256 + threadIdx.x;        // 0 .. 102399
    const int h = idx & (HDIM - 1);
    const float m = stats[h] * (1.0f / BV);
    const float v = stats[HDIM + h] * (1.0f / BV) - m * m;
    const float rs = __frsqrt_rn(v + EPSV);
    const float y = (x_new[idx] - m) * rs * gx[h] + bx[h];
    x_out[idx] = fmaxf(y, 0.0f);
}

// ---------- K4: batchnorm+relu for e (in place on d_out e region) ----------
__global__ void gnn_k_bn_e(float* __restrict__ eio, const float* __restrict__ stats,
                           const float* __restrict__ ge, const float* __restrict__ be) {
    const size_t idx0 = ((size_t)blockIdx.x * 256 + threadIdx.x) * 8;
    float4* p = (float4*)(eio + idx0);
    float4 v0 = p[0], v1 = p[1];
    float vals[8] = {v0.x, v0.y, v0.z, v0.w, v1.x, v1.y, v1.z, v1.w};
    const int hb = (int)(idx0 & (HDIM - 1));
#pragma unroll
    for (int k = 0; k < 8; ++k) {
        const int h = hb + k;
        const float m  = stats[2 * HDIM + h] * (1.0f / BVV);
        const float vv = stats[3 * HDIM + h] * (1.0f / BVV) - m * m;
        const float rs = __frsqrt_rn(vv + EPSV);
        vals[k] = fmaxf((vals[k] - m) * rs * ge[h] + be[h], 0.0f);
    }
    p[0] = make_float4(vals[0], vals[1], vals[2], vals[3]);
    p[1] = make_float4(vals[4], vals[5], vals[6], vals[7]);
}

// ---------- launcher ----------
extern "C" void kernel_launch(void* const* d_in, const int* in_sizes, int n_in,
                              void* d_out, int out_size, void* d_ws, size_t ws_size,
                              hipStream_t stream) {
    const float* x  = (const float*)d_in[0];
    const float* e  = (const float*)d_in[1];
    // d_in[2] = graph (unused by the reference math)
    const float* Uw = (const float*)d_in[3];
    const float* Ub = (const float*)d_in[4];
    const float* Vw = (const float*)d_in[5];
    const float* Vb = (const float*)d_in[6];
    const float* Aw = (const float*)d_in[7];
    const float* Ab = (const float*)d_in[8];
    const float* Bw = (const float*)d_in[9];
    const float* Bb = (const float*)d_in[10];
    const float* Cw = (const float*)d_in[11];
    const float* Cb = (const float*)d_in[12];
    const float* gx = (const float*)d_in[13];
    const float* bx = (const float*)d_in[14];
    const float* ge = (const float*)d_in[15];
    const float* be = (const float*)d_in[16];

    float* out = (float*)d_out;
    float* x_out = out;                         // 800*128
    float* e_out = out + (size_t)BV * HDIM;     // 320000*128 (also e_new scratch)

    // workspace layout (all 256B-aligned)
    char* ws = (char*)d_ws;
    _Float16* Wh   = (_Float16*)ws;                             // 5*128*128 f16 = 160 KB
    float*    stats = (float*)(ws + 163840);                    // 4*128 f32
    float*    xw4   = (float*)(ws + 165888);                    // 4*800*128 f32
    float*    x_new = (float*)(ws + 1804288);                   // 800*128 f32

    gnn_k_init<<<320, 256, 0, stream>>>(Uw, Vw, Aw, Bw, Cw, Wh, stats);
    gnn_k_lin4<<<dim3(50, 4), 256, 0, stream>>>(x, Wh, Ub, Vb, Ab, Bb, xw4);
    gnn_k_edge<<<BV, 256, 0, stream>>>(e, Wh + 4 * (HDIM * HDIM), Cb, xw4,
                                       e_out, x_new, stats);
    gnn_k_bn_x<<<(BV * HDIM) / 256, 256, 0, stream>>>(x_new, stats, gx, bx, x_out);
    gnn_k_bn_e<<<((size_t)BVV * HDIM) / (256 * 8), 256, 0, stream>>>(e_out, stats, ge, be);
}
